// RWKVCellV4_58617713656063
// MI455X (gfx1250) — compile-verified, matrix operand order checked by reading, not run on hardware
//
#include <hip/hip_runtime.h>
#include <hip/hip_bf16.h>

// ---------------------------------------------------------------------------
// RWKV cell, MI455X (gfx1250, wave32).
//   kernel 0 (x5): f32 -> bf16 pre-convert of x and all 4 weights into d_ws
//   kernel 1: r/k/v = x @ {w_r,w_k,w_v}^T  (bf16 WMMA, f32 accum)
//             async global->LDS tile fills (ASYNCcnt, inline asm),
//             double buffered, fused stabilized-WKV elementwise epilogue.
//   kernel 2: output = y @ w_o^T (bf16 WMMA) -> d_out[0..B*D)
// ---------------------------------------------------------------------------

#define BDIM 4096
#define DDIM 2048
#define BM 64        // rows per block
#define BN 128       // cols per block
#define KC 32        // K chunk (bf16 WMMA depth)
#define KPAD 40      // LDS row stride in bf16 elements (80B, bank spread)

typedef __attribute__((ext_vector_type(16))) __bf16 v16bf;
typedef __attribute__((ext_vector_type(8)))  __bf16 v8bf;
typedef __attribute__((ext_vector_type(4)))  __bf16 v4bf;
typedef __attribute__((ext_vector_type(8)))  float  v8f;

__device__ __forceinline__ __bf16 f2bf(float f) {
  union { float f; unsigned u; } c; c.f = f;
  unsigned r = c.u + 0x7FFFu + ((c.u >> 16) & 1u);   // round-to-nearest-even
  unsigned short h = (unsigned short)(r >> 16);
  __bf16 out;
  __builtin_memcpy(&out, &h, sizeof(h));
  return out;
}

// ---- CDNA5 async global->LDS copy (ASYNCcnt) via inline asm ----------------
// ISA 15.18.3: GLOBAL_LOAD_ASYNC_TO_LDS_B128  vdst(LDS byte addr), vaddr64, off
// Generic LDS pointers carry the DS offset in addr[31:0] (aperture is in the
// high half), so a 32-bit truncation yields the LDS byte address.
#if defined(__AMDGCN__)
#define HAVE_ASYNC_LDS 1
#else
#define HAVE_ASYNC_LDS 0
#endif

__device__ __forceinline__ void async_copy16(const __bf16* g, __bf16* l) {
#if HAVE_ASYNC_LDS
  unsigned loff = (unsigned)(size_t)l;
  asm volatile("global_load_async_to_lds_b128 %0, %1, off"
               :
               : "v"(loff), "v"(g)
               : "memory");
#else
  *(v8bf*)l = *(const v8bf*)g;
#endif
}

__device__ __forceinline__ void wait_async_lds() {
#if HAVE_ASYNC_LDS
  asm volatile("s_wait_asynccnt 0x0" ::: "memory");
#endif
}

// Fragment loader for V_WMMA_F32_16X16X32_BF16 A/B operands.
// Lanes 0-15 hold K {0..7, 16..23}; lanes 16-31 hold K {8..15, 24..31}.
__device__ __forceinline__ v16bf load_frag(const __bf16* rowptr, int lane) {
  const int hi = (lane >> 4) & 1;
  v8bf lo8 = *(const v8bf*)(rowptr + hi * 8);
  v8bf hi8 = *(const v8bf*)(rowptr + 16 + hi * 8);
  v16bf r;
  *((v8bf*)&r)       = lo8;
  *(((v8bf*)&r) + 1) = hi8;
  return r;
}

// ------------------------------ kernel 0 -----------------------------------
__global__ __launch_bounds__(256) void cvt_bf16_kernel(
    const float* __restrict__ src, __bf16* __restrict__ dst, int n4) {
  const int i = blockIdx.x * blockDim.x + threadIdx.x;
  if (i < n4) {
    const float4 f = ((const float4*)src)[i];
    v4bf b;
    b[0] = f2bf(f.x); b[1] = f2bf(f.y); b[2] = f2bf(f.z); b[3] = f2bf(f.w);
    ((v4bf*)dst)[i] = b;
  }
}

// ------------------------------ kernel 1 -----------------------------------
__global__ __launch_bounds__(256) void rwkv_proj_kernel(
    const __bf16* __restrict__ xb,    const float* __restrict__ frac_n,
    const float* __restrict__ frac_d, const float* __restrict__ scale,
    const __bf16* __restrict__ wrb,   const __bf16* __restrict__ wkb,
    const __bf16* __restrict__ wvb,   const float* __restrict__ w_u,
    const float* __restrict__ w_w,    float* __restrict__ out,
    __bf16* __restrict__ ybuf)
{
  __shared__ __align__(16) __bf16 sx[2][BM][KPAD];
  __shared__ __align__(16) __bf16 sw[2][3][BN][KPAD];

  const int tid  = threadIdx.x;
  const int lane = tid & 31;
  const int wid  = tid >> 5;
  const int wm   = wid >> 2;                 // 0..1   (32-row slab)
  const int wn   = wid & 3;                  // 0..3   (32-col slab)
  const int rowBase = blockIdx.y * BM;
  const int colBase = blockIdx.x * BN;

  v8f acc[3][2][2] = {};                     // [proj][mTile][nTile]

  auto load_chunk = [&](int kc, int bb) {
    // x tile: 64 rows x 32 bf16 = 256 x 16B, one per thread
    {
      const int seg = tid & 3, row = tid >> 2;
      async_copy16(xb + (size_t)(rowBase + row) * DDIM + kc + seg * 8,
                   &sx[bb][row][seg * 8]);
    }
    // weight tiles: 3 x 128 rows x 32 bf16 = 1536 x 16B, six per thread
#pragma unroll
    for (int i = 0; i < 6; ++i) {
      const int idx = tid + i * 256;
      const int seg = idx & 3;
      const int row = (idx >> 2) & 127;
      const int p   = idx >> 9;              // 0..2
      const __bf16* wp = (p == 0) ? wrb : ((p == 1) ? wkb : wvb);
      const __bf16* gp = wp + (size_t)(colBase + row) * DDIM + kc + seg * 8;
      if (i == 0 && kc + 2 * KC < DDIM)
        __builtin_prefetch(gp + 2 * KC, 0, 1);   // global_prefetch_b8
      async_copy16(gp, &sw[bb][p][row][seg * 8]);
    }
  };

  load_chunk(0, 0);

  for (int kc = 0; kc < DDIM; kc += KC) {
    const int bb = (kc / KC) & 1;
    wait_async_lds();
    __syncthreads();
    if (kc + KC < DDIM) load_chunk(kc + KC, bb ^ 1);

    v16bf a[2], b[3][2];
#pragma unroll
    for (int m = 0; m < 2; ++m)
      a[m] = load_frag(&sx[bb][wm * 32 + m * 16 + (lane & 15)][0], lane);
#pragma unroll
    for (int p = 0; p < 3; ++p)
#pragma unroll
      for (int n = 0; n < 2; ++n)
        b[p][n] = load_frag(&sw[bb][p][wn * 32 + n * 16 + (lane & 15)][0], lane);

#pragma unroll
    for (int p = 0; p < 3; ++p)
#pragma unroll
      for (int m = 0; m < 2; ++m)
#pragma unroll
        for (int n = 0; n < 2; ++n)
          acc[p][m][n] = __builtin_amdgcn_wmma_f32_16x16x32_bf16(
              false, a[m], false, b[p][n], (short)0, acc[p][m][n],
              false, false);
  }

  // --------- fused WKV elementwise epilogue -------------------------------
#pragma unroll
  for (int m = 0; m < 2; ++m) {
#pragma unroll
    for (int n = 0; n < 2; ++n) {
      const int col = colBase + wn * 32 + n * 16 + (lane & 15);
      const float wu = w_u[col];
      const float ww = w_w[col];
#pragma unroll
      for (int i = 0; i < 8; ++i) {
        const int row =
            rowBase + wm * 32 + m * 16 + ((lane < 16) ? i : i + 8);
        const size_t off = (size_t)row * DDIM + col;
        const float rv = acc[0][m][n][i];
        const float kv = acc[1][m][n][i];
        const float vv = acc[2][m][n][i];
        const float sc = scale[off];
        const float fn = frac_n[off];
        const float fd = frac_d[off];

        const float rs  = 1.f / (1.f + __expf(-rv));
        // output path (with bonus w_u)
        const float sms = fmaxf(sc, wu + kv);
        const float cm  = __expf(sc - sms);
        const float am  = __expf(wu + kv - sms);
        const float fnt = cm * fn + am * vv;
        const float fdt = cm * fd + am;
        const float y   = rs * fnt / fdt;
        // state path (decay w_w, no bonus)
        const float ns  = fmaxf(sc + ww, kv);
        const float ncm = __expf(sc + ww - ns);
        const float nam = __expf(kv - ns);

        ybuf[off] = f2bf(y);
        out[(size_t)1 * BDIM * DDIM + off] = ncm * fn + nam * vv;
        out[(size_t)2 * BDIM * DDIM + off] = ncm * fd + nam;
        out[(size_t)3 * BDIM * DDIM + off] = ns;
      }
    }
  }
}

// ------------------------------ kernel 2 -----------------------------------
__global__ __launch_bounds__(256) void rwkv_out_kernel(
    const __bf16* __restrict__ y, const __bf16* __restrict__ wob,
    float* __restrict__ out)
{
  __shared__ __align__(16) __bf16 sy[2][BM][KPAD];
  __shared__ __align__(16) __bf16 sw[2][BN][KPAD];

  const int tid  = threadIdx.x;
  const int lane = tid & 31;
  const int wid  = tid >> 5;
  const int wm   = wid >> 2;
  const int wn   = wid & 3;
  const int rowBase = blockIdx.y * BM;
  const int colBase = blockIdx.x * BN;

  v8f acc[2][2] = {};

  auto load_chunk = [&](int kc, int bb) {
    // y tile: 64 x 32 bf16 = 256 x 16B, one per thread
    {
      const int seg = tid & 3, row = tid >> 2;
      async_copy16(y + (size_t)(rowBase + row) * DDIM + kc + seg * 8,
                   &sy[bb][row][seg * 8]);
    }
    // w_o tile: 128 x 32 bf16 = 512 x 16B, two per thread
#pragma unroll
    for (int i = 0; i < 2; ++i) {
      const int idx = tid + i * 256;
      const int seg = idx & 3, row = idx >> 2;
      const __bf16* gp = wob + (size_t)(colBase + row) * DDIM + kc + seg * 8;
      if (i == 0 && kc + 2 * KC < DDIM)
        __builtin_prefetch(gp + 2 * KC, 0, 1);
      async_copy16(gp, &sw[bb][row][seg * 8]);
    }
  };

  load_chunk(0, 0);

  for (int kc = 0; kc < DDIM; kc += KC) {
    const int bb = (kc / KC) & 1;
    wait_async_lds();
    __syncthreads();
    if (kc + KC < DDIM) load_chunk(kc + KC, bb ^ 1);

    v16bf a[2], b[2];
#pragma unroll
    for (int m = 0; m < 2; ++m)
      a[m] = load_frag(&sy[bb][wm * 32 + m * 16 + (lane & 15)][0], lane);
#pragma unroll
    for (int n = 0; n < 2; ++n)
      b[n] = load_frag(&sw[bb][wn * 32 + n * 16 + (lane & 15)][0], lane);

#pragma unroll
    for (int m = 0; m < 2; ++m)
#pragma unroll
      for (int n = 0; n < 2; ++n)
        acc[m][n] = __builtin_amdgcn_wmma_f32_16x16x32_bf16(
            false, a[m], false, b[n], (short)0, acc[m][n], false, false);
  }

#pragma unroll
  for (int m = 0; m < 2; ++m)
#pragma unroll
    for (int n = 0; n < 2; ++n) {
      const int col = colBase + wn * 32 + n * 16 + (lane & 15);
#pragma unroll
      for (int i = 0; i < 8; ++i) {
        const int row =
            rowBase + wm * 32 + m * 16 + ((lane < 16) ? i : i + 8);
        out[(size_t)row * DDIM + col] = acc[m][n][i];
      }
    }
}

// ------------------------------ launch -------------------------------------
extern "C" void kernel_launch(void* const* d_in, const int* in_sizes, int n_in,
                              void* d_out, int out_size, void* d_ws, size_t ws_size,
                              hipStream_t stream) {
  const float* x      = (const float*)d_in[0];
  const float* frac_n = (const float*)d_in[1];
  const float* frac_d = (const float*)d_in[2];
  const float* scale  = (const float*)d_in[3];
  const float* w_r    = (const float*)d_in[4];
  const float* w_k    = (const float*)d_in[5];
  const float* w_v    = (const float*)d_in[6];
  const float* w_o    = (const float*)d_in[7];
  const float* w_u    = (const float*)d_in[8];
  const float* w_w    = (const float*)d_in[9];
  float* out = (float*)d_out;

  // ws layout (bf16 elements): x | w_r | w_k | w_v | w_o | y   == 64 MiB
  const size_t BD = (size_t)BDIM * DDIM;   // 8.39M
  const size_t DD = (size_t)DDIM * DDIM;   // 4.19M
  __bf16* xb   = (__bf16*)d_ws;
  __bf16* wrb  = xb + BD;
  __bf16* wkb  = wrb + DD;
  __bf16* wvb  = wkb + DD;
  __bf16* wob  = wvb + DD;
  __bf16* ybuf = wob + DD;

  // kernel 0: pre-convert to bf16 (bandwidth-bound)
  const int n4x = (int)(BD / 4), n4w = (int)(DD / 4);
  cvt_bf16_kernel<<<n4x / 256, 256, 0, stream>>>(x,   xb,  n4x);
  cvt_bf16_kernel<<<n4w / 256, 256, 0, stream>>>(w_r, wrb, n4w);
  cvt_bf16_kernel<<<n4w / 256, 256, 0, stream>>>(w_k, wkb, n4w);
  cvt_bf16_kernel<<<n4w / 256, 256, 0, stream>>>(w_v, wvb, n4w);
  cvt_bf16_kernel<<<n4w / 256, 256, 0, stream>>>(w_o, wob, n4w);

  dim3 grid(DDIM / BN, BDIM / BM);   // (16, 64)
  rwkv_proj_kernel<<<grid, 256, 0, stream>>>(xb, frac_n, frac_d, scale,
                                             wrb, wkb, wvb, w_u, w_w,
                                             out, ybuf);
  rwkv_out_kernel<<<grid, 256, 0, stream>>>(ybuf, wob, out);
}
